// BasicAttentionBlock_32839319945425
// MI455X (gfx1250) — matrix operand order repaired and checked by reading, not compiled, not run on hardware
//
#include <hip/hip_runtime.h>
#include <hip/hip_bf16.h>

// ---------------------------------------------------------------------------
// MI455X (gfx1250) fused attention block, bf16 WMMA (fp32 accum).
// Compute-bound (~37 GFLOP vs ~50MB traffic) -> v_wmma_f32_16x16x32_bf16.
// K/V tiles staged to LDS with GLOBAL_LOAD_ASYNC_TO_LDS_B128, double-buffered.
// ---------------------------------------------------------------------------

typedef __bf16 bf16_t;
typedef __attribute__((ext_vector_type(16))) __bf16       v16bf;
typedef __attribute__((ext_vector_type(8)))  float        v8f;
typedef __attribute__((ext_vector_type(4)))  unsigned int uint4v;

constexpr int kC      = 128;           // channels
constexpr int kHW     = 4096;          // 64*64 pixels
constexpr int kB      = 4;             // batch
constexpr int kGroups = 8;
constexpr int kGsz    = kC / kGroups;  // 16 channels per group
constexpr float kEps  = 1e-5f;
constexpr float kScale = 0.08838834764831845f;  // 1/sqrt(128)

constexpr int kBQ = 64;   // query rows per block (flash)
constexpr int kBK = 64;   // key rows per iteration

union frag16u {
  v16bf bf;
  uint4v u[2];
};

// ---------------------------------------------------------------------------
// Async global->LDS (CDNA5): per-lane 16B copy, tracked by ASYNCcnt.
// ---------------------------------------------------------------------------
__device__ __forceinline__ void async_load_b128(void* lds_ptr, const void* gptr) {
  unsigned lds_off = (unsigned)(uintptr_t)lds_ptr;  // low 32 bits = LDS byte offset
  unsigned long long ga = (unsigned long long)(uintptr_t)gptr;
  asm volatile("global_load_async_to_lds_b128 %0, %1, off"
               :: "v"(lds_off), "v"(ga)
               : "memory");
}
__device__ __forceinline__ void wait_async0() {
  asm volatile("s_wait_asynccnt 0x0" ::: "memory");
}

// ---------------------------------------------------------------------------
// WMMA fragment helpers (wave32; A/B use contiguous k = 16h+s here, C/D per CDNA5 ISA 7.12.2)
// ---------------------------------------------------------------------------

__device__ __forceinline__ v8f wmma_bf16(v16bf a, v16bf b, v8f c) {
  return __builtin_amdgcn_wmma_f32_16x16x32_bf16(
      /*neg_a=*/false, a, /*neg_b=*/false, b,
      /*c_mod=*/(short)0, c, /*reuse_a=*/false, /*reuse_b=*/false);
}

// A fragment: 16x32 bf16 tile, row-major at src, leading dim ld (elements).
__device__ __forceinline__ v16bf load_a_frag(const bf16_t* src, int ld) {
  int lane = threadIdx.x & 31;
  int m = lane & 15, hh = lane >> 4;
  const bf16_t* p = src + (size_t)m * ld + hh * 16;
  frag16u f;
  f.u[0] = *(const uint4v*)(p);
  f.u[1] = *(const uint4v*)(p + 8);
  return f.bf;
}

// B fragment from transposed storage: memory holds T[n][k] row-major
// (logical B[k][n] = T[n][k]).  Per lane: 16 contiguous elems at n*ld+h*16.
__device__ __forceinline__ v16bf load_bT_frag(const bf16_t* src, int ld) {
  int lane = threadIdx.x & 31;
  int n = lane & 15, hh = lane >> 4;
  const bf16_t* p = src + (size_t)n * ld + hh * 16;
  frag16u f;
  f.u[0] = *(const uint4v*)(p);
  f.u[1] = *(const uint4v*)(p + 8);
  return f.bf;
}

// B fragment from pre-swizzled weights: lane's 16 elems at tile + lane*16.
__device__ __forceinline__ v16bf load_b_swz(const bf16_t* tile) {
  int lane = threadIdx.x & 31;
  const bf16_t* p = tile + lane * 16;
  frag16u f;
  f.u[0] = *(const uint4v*)(p);
  f.u[1] = *(const uint4v*)(p + 8);
  return f.bf;
}

// C/D 16x16 f32: VGPR r -> M = r + 8*(lane>=16), N = lane&15.
__device__ __forceinline__ void store_d_f32(float* dst, int ld, v8f d) {
  int lane = threadIdx.x & 31;
  int n = lane & 15, hh = lane >> 4;
#pragma unroll
  for (int r = 0; r < 8; ++r) dst[(size_t)(r + hh * 8) * ld + n] = d[r];
}

__device__ __forceinline__ void store_d_bf16(bf16_t* dst, int ld, v8f d) {
  int lane = threadIdx.x & 31;
  int n = lane & 15, hh = lane >> 4;
#pragma unroll
  for (int r = 0; r < 8; ++r) dst[(size_t)(r + hh * 8) * ld + n] = (bf16_t)d[r];
}

// ---------------------------------------------------------------------------
// Kernel 1: GroupNorm stats (mean, rstd) per (batch, group)
// ---------------------------------------------------------------------------
__global__ __launch_bounds__(256) void gn_stats_kernel(
    const float* __restrict__ x, float* __restrict__ stats) {
  __shared__ float ssum[256], ssq[256];
  int b = blockIdx.x / kGroups, g = blockIdx.x % kGroups;
  const float* p = x + ((size_t)b * kC + (size_t)g * kGsz) * kHW;
  const int N = kGsz * kHW;  // 65536
  float s = 0.f, q = 0.f;
  for (int i = threadIdx.x; i < N; i += 256) {
    float v = p[i];
    s += v;
    q += v * v;
  }
  ssum[threadIdx.x] = s;
  ssq[threadIdx.x] = q;
  __syncthreads();
  for (int off = 128; off > 0; off >>= 1) {
    if ((int)threadIdx.x < off) {
      ssum[threadIdx.x] += ssum[threadIdx.x + off];
      ssq[threadIdx.x]  += ssq[threadIdx.x + off];
    }
    __syncthreads();
  }
  if (threadIdx.x == 0) {
    float mean = ssum[0] / (float)N;
    float var  = ssq[0] / (float)N - mean * mean;
    stats[blockIdx.x * 2 + 0] = mean;
    stats[blockIdx.x * 2 + 1] = rsqrtf(var + kEps);
  }
}

// ---------------------------------------------------------------------------
// Kernel 2: GN apply + transpose [B,C,HW] f32 -> [B,HW,C] bf16
// ---------------------------------------------------------------------------
__global__ __launch_bounds__(256) void gn_apply_kernel(
    const float* __restrict__ x, const float* __restrict__ gnw,
    const float* __restrict__ gnb, const float* __restrict__ stats,
    bf16_t* __restrict__ h) {
  int idx = blockIdx.x * 256 + threadIdx.x;  // over B*C*HW = 2M
  int p = idx % kHW;
  int c = (idx / kHW) % kC;
  int b = idx / (kHW * kC);
  int sg = b * kGroups + c / kGsz;
  float mean = stats[sg * 2 + 0];
  float rstd = stats[sg * 2 + 1];
  float v = (x[idx] - mean) * rstd * gnw[c] + gnb[c];
  h[((size_t)b * kHW + p) * kC + c] = (bf16_t)v;
}

// ---------------------------------------------------------------------------
// Kernel 3: weight cast + swizzle to fragment-ready layout.
// ---------------------------------------------------------------------------
__global__ __launch_bounds__(256) void wtrans_kernel(
    const float* __restrict__ wq, const float* __restrict__ wk,
    const float* __restrict__ wv, const float* __restrict__ wp,
    bf16_t* __restrict__ tq, bf16_t* __restrict__ tk,
    bf16_t* __restrict__ tv, bf16_t* __restrict__ tp) {
  int idx = blockIdx.x * 256 + threadIdx.x;  // over 4*C*C
  int sel = idx / (kC * kC);
  int r = idx % (kC * kC);
  int co = r / kC, ci = r % kC;
  const float* src = (sel == 0) ? wq : (sel == 1) ? wk : (sel == 2) ? wv : wp;
  bf16_t* dst      = (sel == 0) ? tq : (sel == 1) ? tk : (sel == 2) ? tv : tp;
  int kt = ci >> 5, half = (ci >> 4) & 1, v = ci & 15;
  int nt = co >> 4, n = co & 15;
  int lane = n | (half << 4);
  dst[((kt * 8 + nt) * 32 + lane) * 16 + v] = (bf16_t)src[co * kC + ci];
}

// ---------------------------------------------------------------------------
// Kernel 4: GEMM  out[pix][co] = sum_ci h[pix][ci]*W[co][ci] + bias[co]
// grid (HW/128, C/64, B); 8 waves, each a 32x32 tile (2x2 WMMA tiles).
// ---------------------------------------------------------------------------
__global__ __launch_bounds__(256) void qkv_gemm_kernel(
    const bf16_t* __restrict__ h, const bf16_t* __restrict__ wswz,
    const float* __restrict__ bias, bf16_t* __restrict__ out) {
  int b = blockIdx.z;
  int row0 = blockIdx.x * 128;
  int col0 = blockIdx.y * 64;
  int w = threadIdx.x >> 5;
  int r0 = row0 + (w >> 1) * 32;
  int c0 = col0 + (w & 1) * 32;
  int lane = threadIdx.x & 31;
  int n = lane & 15;

  const bf16_t* A = h + (size_t)b * kHW * kC;
  v8f acc[2][2];
#pragma unroll
  for (int i = 0; i < 2; ++i)
#pragma unroll
    for (int j = 0; j < 2; ++j) {
      float bv = bias[c0 + j * 16 + n];
#pragma unroll
      for (int r = 0; r < 8; ++r) acc[i][j][r] = bv;
    }

#pragma unroll
  for (int k0 = 0; k0 < kC; k0 += 32) {
    v16bf a0 = load_a_frag(A + (size_t)r0 * kC + k0, kC);
    v16bf a1 = load_a_frag(A + (size_t)(r0 + 16) * kC + k0, kC);
    int tb = (k0 >> 5) * 8 + (c0 >> 4);
    v16bf b0 = load_b_swz(wswz + (size_t)tb * 512);
    v16bf b1 = load_b_swz(wswz + (size_t)(tb + 1) * 512);
    acc[0][0] = wmma_bf16(a0, b0, acc[0][0]);
    acc[0][1] = wmma_bf16(a0, b1, acc[0][1]);
    acc[1][0] = wmma_bf16(a1, b0, acc[1][0]);
    acc[1][1] = wmma_bf16(a1, b1, acc[1][1]);
  }

  bf16_t* O = out + (size_t)b * kHW * kC;
#pragma unroll
  for (int i = 0; i < 2; ++i)
#pragma unroll
    for (int j = 0; j < 2; ++j)
      store_d_bf16(O + (size_t)(r0 + i * 16) * kC + (c0 + j * 16), kC, acc[i][j]);
}

// ---------------------------------------------------------------------------
// Kernel 4b: V projection writing channel-major vT[B][C][HW] bf16.
// D-layout: per lane (fixed N=cout) the 8 accum elems are 8 consecutive
// pixels -> one packed 16-byte store per lane.
// ---------------------------------------------------------------------------
__global__ __launch_bounds__(256) void v_gemm_kernel(
    const bf16_t* __restrict__ h, const bf16_t* __restrict__ wswz,
    const float* __restrict__ bias, bf16_t* __restrict__ vT) {
  int b = blockIdx.z;
  int row0 = blockIdx.x * 128;
  int col0 = blockIdx.y * 64;
  int w = threadIdx.x >> 5;
  int r0 = row0 + (w >> 1) * 32;
  int c0 = col0 + (w & 1) * 32;
  int lane = threadIdx.x & 31;
  int n = lane & 15, hh = lane >> 4;

  const bf16_t* A = h + (size_t)b * kHW * kC;
  v8f acc[2][2];
#pragma unroll
  for (int i = 0; i < 2; ++i)
#pragma unroll
    for (int j = 0; j < 2; ++j) {
      float bv = bias[c0 + j * 16 + n];
#pragma unroll
      for (int r = 0; r < 8; ++r) acc[i][j][r] = bv;
    }

#pragma unroll
  for (int k0 = 0; k0 < kC; k0 += 32) {
    v16bf a0 = load_a_frag(A + (size_t)r0 * kC + k0, kC);
    v16bf a1 = load_a_frag(A + (size_t)(r0 + 16) * kC + k0, kC);
    int tb = (k0 >> 5) * 8 + (c0 >> 4);
    v16bf b0 = load_b_swz(wswz + (size_t)tb * 512);
    v16bf b1 = load_b_swz(wswz + (size_t)(tb + 1) * 512);
    acc[0][0] = wmma_bf16(a0, b0, acc[0][0]);
    acc[0][1] = wmma_bf16(a0, b1, acc[0][1]);
    acc[1][0] = wmma_bf16(a1, b0, acc[1][0]);
    acc[1][1] = wmma_bf16(a1, b1, acc[1][1]);
  }

  bf16_t* O = vT + (size_t)b * kC * kHW;
#pragma unroll
  for (int i = 0; i < 2; ++i)
#pragma unroll
    for (int j = 0; j < 2; ++j) {
      int ng = c0 + j * 16 + n;        // output channel
      int pix0 = r0 + i * 16 + hh * 8; // 8 consecutive pixels per lane
      union { uint4v u; bf16_t e[8]; } t;
#pragma unroll
      for (int r = 0; r < 8; ++r) t.e[r] = (bf16_t)acc[i][j][r];
      *(uint4v*)(O + (size_t)ng * kHW + pix0) = t.u;
    }
}

// ---------------------------------------------------------------------------
// Kernel 5: flash attention. 64 query rows/block, 64-key tiles double-
// buffered through LDS via async global->LDS loads (ASYNCcnt).
// ---------------------------------------------------------------------------
__global__ __launch_bounds__(256) void attn_kernel(
    const bf16_t* __restrict__ qg, const bf16_t* __restrict__ kg,
    const bf16_t* __restrict__ vTg, bf16_t* __restrict__ og) {
  __shared__ __align__(16) bf16_t Qs[kBQ][kC];        // 16 KB
  __shared__ __align__(16) bf16_t Ks[2][kBK][kC];     // 32 KB
  __shared__ __align__(16) bf16_t VsT[2][kC][kBK];    // 32 KB (channel-major)
  __shared__ __align__(16) float  Ss[kBQ][kBK];       // 16 KB
  __shared__ __align__(16) bf16_t Ps[kBQ][kBK];       //  8 KB
  __shared__ float mrow[kBQ], lrow[kBQ], corr[kBQ];

  int b  = blockIdx.y;
  int q0 = blockIdx.x * kBQ;
  int tid = threadIdx.x;
  int w = tid >> 5;
  int lane = tid & 31;
  int nn = lane & 15, hh = lane >> 4;

  const bf16_t* Q  = qg  + (size_t)b * kHW * kC;
  const bf16_t* K  = kg  + (size_t)b * kHW * kC;
  const bf16_t* vT = vTg + (size_t)b * kC * kHW;

  // load Q block (16B vectors)
  {
    const uint4v* src = (const uint4v*)(Q + (size_t)q0 * kC);
    uint4v* dst = (uint4v*)(&Qs[0][0]);
    for (int i = tid; i < kBQ * kC / 8; i += 256) dst[i] = src[i];
  }
  if (tid < kBQ) { mrow[tid] = -1e30f; lrow[tid] = 0.f; }

  // persistent output accumulators: wave w owns tiles t = 4w+i in 4x8 grid
  v8f acc[4];
#pragma unroll
  for (int i = 0; i < 4; ++i)
#pragma unroll
    for (int r = 0; r < 8; ++r) acc[i][r] = 0.f;

  // async fill of one K/V tile pair into buffer `bu` for key block j0
  auto fill_async = [&](int bu, int j0) {
    const bf16_t* kp = K + (size_t)j0 * kC;
#pragma unroll
    for (int it = 0; it < 4; ++it) {           // kBK*kC/8/256 = 4
      int i = tid + it * 256;
      int j = i >> 4, cc = (i & 15) * 8;       // 16 chunks per 128-elem row
      async_load_b128(&Ks[bu][j][cc], kp + (size_t)j * kC + cc);
    }
#pragma unroll
    for (int it = 0; it < 4; ++it) {           // kC*kBK/8/256 = 4
      int i = tid + it * 256;
      int c = i >> 3, jc = (i & 7) * 8;        // 8 chunks per 64-elem row
      async_load_b128(&VsT[bu][c][jc], vT + (size_t)c * kHW + j0 + jc);
    }
  };

  int bu = 0;
  fill_async(bu, 0);

  for (int j0 = 0; j0 < kHW; j0 += kBK) {
    wait_async0();        // drain fills of current buffer
    __syncthreads();      // all waves' fills landed; prev-buffer readers done
    if (j0 + kBK < kHW) fill_async(bu ^ 1, j0 + kBK);  // overlap next tile

    // S = scale * Q K^T : 4x4 tiles, two per wave
#pragma unroll
    for (int i = 0; i < 2; ++i) {
      int t = w * 2 + i;
      int tm = t >> 2, tn = t & 3;
      v8f s;
#pragma unroll
      for (int r = 0; r < 8; ++r) s[r] = 0.f;
#pragma unroll
      for (int k0 = 0; k0 < kC; k0 += 32) {
        v16bf a = load_a_frag(&Qs[tm * 16][k0], kC);
        v16bf bb = load_bT_frag(&Ks[bu][tn * 16][k0], kC);
        s = wmma_bf16(a, bb, s);
      }
#pragma unroll
      for (int r = 0; r < 8; ++r) s[r] *= kScale;
      store_d_f32(&Ss[tm * 16][tn * 16], kBK, s);
    }
    __syncthreads();

    // Online softmax: 4 lanes per row, shfl_xor reductions (all 256 threads)
    {
      int r = tid >> 2, sub = tid & 3;
      float m_old = mrow[r];
      float mx = m_old;
      int jb = sub * 16;
#pragma unroll
      for (int j = 0; j < 16; ++j) mx = fmaxf(mx, Ss[r][jb + j]);
      mx = fmaxf(mx, __shfl_xor(mx, 1));
      mx = fmaxf(mx, __shfl_xor(mx, 2));
      float sum = 0.f;
#pragma unroll
      for (int j = 0; j < 16; ++j) {
        float pv = __expf(Ss[r][jb + j] - mx);
        Ps[r][jb + j] = (bf16_t)pv;
        sum += pv;
      }
      sum += __shfl_xor(sum, 1);
      sum += __shfl_xor(sum, 2);
      if (sub == 0) {
        float cf = __expf(m_old - mx);
        mrow[r] = mx;
        lrow[r] = lrow[r] * cf + sum;
        corr[r] = cf;
      }
    }
    __syncthreads();

    // acc = corr(row)*acc + P @ V : 4x8 tiles, four per wave
#pragma unroll
    for (int i = 0; i < 4; ++i) {
      int t = w * 4 + i;
      int tm = t >> 3, tn = t & 7;
#pragma unroll
      for (int r = 0; r < 8; ++r) acc[i][r] *= corr[tm * 16 + r + hh * 8];
#pragma unroll
      for (int k0 = 0; k0 < kBK; k0 += 32) {
        v16bf a = load_a_frag(&Ps[tm * 16][k0], kBK);
        v16bf bb = load_bT_frag(&VsT[bu][tn * 16][k0], kBK);
        acc[i] = wmma_bf16(a, bb, acc[i]);
      }
    }
    bu ^= 1;
  }

  // normalize and write bf16 [B,HW,C]
  bf16_t* O = og + (size_t)b * kHW * kC;
#pragma unroll
  for (int i = 0; i < 4; ++i) {
    int t = w * 4 + i;
    int tm = t >> 3, tn = t & 7;
#pragma unroll
    for (int r = 0; r < 8; ++r) {
      int row = tm * 16 + r + hh * 8;
      float val = acc[i][r] / lrow[row];
      O[(size_t)(q0 + row) * kC + tn * 16 + nn] = (bf16_t)val;
    }
  }
}

// ---------------------------------------------------------------------------
// Kernel 6: output projection + residual, writes fp32 [B,C,HW]
// ---------------------------------------------------------------------------
__global__ __launch_bounds__(256) void proj_kernel(
    const bf16_t* __restrict__ ao, const bf16_t* __restrict__ wswz,
    const float* __restrict__ bias, const float* __restrict__ x,
    float* __restrict__ out) {
  int b = blockIdx.z;
  int row0 = blockIdx.x * 128;
  int col0 = blockIdx.y * 64;
  int w = threadIdx.x >> 5;
  int r0 = row0 + (w >> 1) * 32;
  int c0 = col0 + (w & 1) * 32;
  int lane = threadIdx.x & 31;
  int n = lane & 15, hh = lane >> 4;

  const bf16_t* A = ao + (size_t)b * kHW * kC;
  v8f acc[2][2];
#pragma unroll
  for (int i = 0; i < 2; ++i)
#pragma unroll
    for (int j = 0; j < 2; ++j) {
      float bv = bias[c0 + j * 16 + n];
#pragma unroll
      for (int r = 0; r < 8; ++r) acc[i][j][r] = bv;
    }

#pragma unroll
  for (int k0 = 0; k0 < kC; k0 += 32) {
    v16bf a0 = load_a_frag(A + (size_t)r0 * kC + k0, kC);
    v16bf a1 = load_a_frag(A + (size_t)(r0 + 16) * kC + k0, kC);
    int tb = (k0 >> 5) * 8 + (c0 >> 4);
    v16bf b0 = load_b_swz(wswz + (size_t)tb * 512);
    v16bf b1 = load_b_swz(wswz + (size_t)(tb + 1) * 512);
    acc[0][0] = wmma_bf16(a0, b0, acc[0][0]);
    acc[0][1] = wmma_bf16(a0, b1, acc[0][1]);
    acc[1][0] = wmma_bf16(a1, b0, acc[1][0]);
    acc[1][1] = wmma_bf16(a1, b1, acc[1][1]);
  }

  const float* xr = x + (size_t)b * kC * kHW;
  float* orow = out + (size_t)b * kC * kHW;
#pragma unroll
  for (int i = 0; i < 2; ++i)
#pragma unroll
    for (int j = 0; j < 2; ++j) {
      int ng = c0 + j * 16 + n;  // output channel
#pragma unroll
      for (int r = 0; r < 8; ++r) {
        int pix = r0 + i * 16 + r + hh * 8;
        size_t idx = (size_t)ng * kHW + pix;
        orow[idx] = xr[idx] + acc[i][j][r];
      }
    }
}

// ---------------------------------------------------------------------------
// Host launcher
// ---------------------------------------------------------------------------
extern "C" void kernel_launch(void* const* d_in, const int* in_sizes, int n_in,
                              void* d_out, int out_size, void* d_ws, size_t ws_size,
                              hipStream_t stream) {
  const float* x   = (const float*)d_in[0];
  const float* gnw = (const float*)d_in[1];
  const float* gnb = (const float*)d_in[2];
  const float* wq  = (const float*)d_in[3];
  const float* bq  = (const float*)d_in[4];
  const float* wk  = (const float*)d_in[5];
  const float* bk  = (const float*)d_in[6];
  const float* wv  = (const float*)d_in[7];
  const float* bv  = (const float*)d_in[8];
  const float* wp  = (const float*)d_in[9];
  const float* bp  = (const float*)d_in[10];
  float* out = (float*)d_out;

  // workspace carve-up (~21 MB total)
  char* ws = (char*)d_ws;
  size_t off = 0;
  auto carve = [&](size_t bytes) -> void* {
    void* p = ws + off;
    off = (off + bytes + 255) & ~(size_t)255;
    return p;
  };
  float*  stats = (float*)carve((size_t)kB * kGroups * 2 * sizeof(float));
  bf16_t* tq = (bf16_t*)carve((size_t)kC * kC * sizeof(bf16_t));
  bf16_t* tk = (bf16_t*)carve((size_t)kC * kC * sizeof(bf16_t));
  bf16_t* tv = (bf16_t*)carve((size_t)kC * kC * sizeof(bf16_t));
  bf16_t* tp = (bf16_t*)carve((size_t)kC * kC * sizeof(bf16_t));
  size_t act = (size_t)kB * kHW * kC * sizeof(bf16_t);  // 4 MB each
  bf16_t* hbuf = (bf16_t*)carve(act);
  bf16_t* qbuf = (bf16_t*)carve(act);
  bf16_t* kbuf = (bf16_t*)carve(act);
  bf16_t* vTbuf = (bf16_t*)carve(act);  // channel-major [B][C][HW]
  bf16_t* abuf = (bf16_t*)carve(act);

  gn_stats_kernel<<<kB * kGroups, 256, 0, stream>>>(x, stats);
  gn_apply_kernel<<<(kB * kC * kHW) / 256, 256, 0, stream>>>(x, gnw, gnb, stats, hbuf);
  wtrans_kernel<<<(4 * kC * kC) / 256, 256, 0, stream>>>(wq, wk, wv, wp, tq, tk, tv, tp);

  dim3 gg(kHW / 128, kC / 64, kB);
  qkv_gemm_kernel<<<gg, 256, 0, stream>>>(hbuf, tq, bq, qbuf);
  qkv_gemm_kernel<<<gg, 256, 0, stream>>>(hbuf, tk, bk, kbuf);
  v_gemm_kernel<<<gg, 256, 0, stream>>>(hbuf, tv, bv, vTbuf);

  attn_kernel<<<dim3(kHW / kBQ, kB), 256, 0, stream>>>(qbuf, kbuf, vTbuf, abuf);

  proj_kernel<<<gg, 256, 0, stream>>>(abuf, tp, bp, x, out);
}
